// DiarizeGPT_83081847374500
// MI455X (gfx1250) — compile-verified, matrix operand order checked by reading, not run on hardware
//
#include <hip/hip_runtime.h>
#include <hip/hip_bf16.h>

// ---------------- model constants ----------------
#define DD   768
#define HH   12
#define DH   64
#define NLAY 8
#define VV   128
#define LAT  1024
#define TAUD 1000
#define LL_  90
#define SS   1120
#define BB   8
#define NTOK (BB*SS)       // 8960
#define QT_PER_BH 70       // S/16
#define QT_PER_BLK 8
#define NQB  9             // ceil(70/8)

// ---------------- WMMA / vector types ----------------
typedef __attribute__((ext_vector_type(16))) __bf16 v16bf;
typedef __attribute__((ext_vector_type(8)))  float  v8f;
typedef unsigned int u32x4 __attribute__((ext_vector_type(4)));
typedef int          i32x4 __attribute__((ext_vector_type(4)));
typedef int          i32x8 __attribute__((ext_vector_type(8)));

union HFrag { v16bf v; unsigned u[8]; unsigned short h[16]; };
union CFrag { v8f v; float f[8]; };

__device__ inline unsigned short f2bf(float f) {
  union { float f; unsigned u; } c; c.f = f;
  unsigned u = c.u + 0x7FFFu + ((c.u >> 16) & 1u);   // round-to-nearest-even
  return (unsigned short)(u >> 16);
}

// Load one 16x32 bf16 A/B fragment (ISA 7.12.2 16-bit layout) from a row-major
// bf16 tile (LDS or global), row stride ldk halves. Lane l: row = l&15,
// K-halves = [g*8 .. g*8+7] and [16+g*8 .. 16+g*8+7]  (g = l>>4).
__device__ inline v16bf load_frag16(const unsigned short* base, int row0, int ldk) {
  int lane = threadIdx.x & 31;
  int g = lane >> 4, m = lane & 15;
  const unsigned short* p = base + (size_t)(row0 + m) * ldk;
  HFrag a;
  const unsigned* p0 = (const unsigned*)(p + g * 8);
  const unsigned* p1 = (const unsigned*)(p + 16 + g * 8);
  a.u[0] = p0[0]; a.u[1] = p0[1]; a.u[2] = p0[2]; a.u[3] = p0[3];
  a.u[4] = p1[0]; a.u[5] = p1[1]; a.u[6] = p1[2]; a.u[7] = p1[3];
  return a.v;
}

__device__ inline v8f wmma_bf16(v16bf a, v16bf b, v8f c) {
  return __builtin_amdgcn_wmma_f32_16x16x32_bf16(false, a, false, b, (short)0, c, false, false);
}

// ---------------- Tensor Data Mover (TDM) 2D tile load ----------------
// Builds a D# (ISA ch.8): group0 = count/lds_addr/global_addr/type, group1 =
// data_size=2B, tensor dims (OOB -> zero fill), tile dims, dim0 stride.
// Tile is written contiguously row-major into LDS.
#if __has_builtin(__builtin_amdgcn_tensor_load_to_lds)
#define USE_TDM 1
__device__ inline void tdm_load_2d(unsigned lds_off, const unsigned short* gptr,
                                   unsigned tensor_d0, unsigned tensor_d1,
                                   unsigned tile_d0, unsigned tile_d1,
                                   unsigned long long stride0) {
  unsigned long long ga = (unsigned long long)(size_t)gptr;
  u32x4 g0;
  g0[0] = 1u;                                            // count=1, user D#
  g0[1] = lds_off;                                       // lds_addr (bytes)
  g0[2] = (unsigned)(ga & 0xFFFFFFFFu);                  // global_addr[31:0]
  g0[3] = (unsigned)((ga >> 32) & 0x01FFFFFFu) | (2u << 30);  // addr[56:32], type=2
  i32x8 g1;
  g1[0] = (int)(1u << 16);                               // data_size=1 -> 2 bytes
  g1[1] = (int)((tensor_d0 & 0xFFFFu) << 16);            // tensor_dim0[15:0]
  g1[2] = (int)(((tensor_d0 >> 16) & 0xFFFFu) | ((tensor_d1 & 0xFFFFu) << 16));
  g1[3] = (int)(((tensor_d1 >> 16) & 0xFFFFu) | ((tile_d0 & 0xFFFFu) << 16));
  g1[4] = (int)(tile_d1 & 0xFFFFu);                      // tile_dim1 (tile_dim2=0)
  g1[5] = (int)(unsigned)(stride0 & 0xFFFFFFFFu);        // dim0 stride[31:0]
  g1[6] = (int)(unsigned)((stride0 >> 32) & 0xFFFFu);    // dim0 stride[47:32]
  g1[7] = 0;
  i32x4 z4 = {0, 0, 0, 0};
#if defined(__clang_major__) && (__clang_major__ >= 23)
  i32x8 z8 = {0, 0, 0, 0, 0, 0, 0, 0};
  __builtin_amdgcn_tensor_load_to_lds(g0, g1, z4, z4, z8, 0);
#else
  __builtin_amdgcn_tensor_load_to_lds(g0, g1, z4, z4, 0);
#endif
}
#else
#define USE_TDM 0
#endif

// ---------------- block reduction helper ----------------
__device__ inline float block_sum256(float v) {
  for (int m = 16; m > 0; m >>= 1) v += __shfl_xor(v, m, 32);
  __shared__ float red[8];
  __shared__ float total;
  int w = threadIdx.x >> 5;
  if ((threadIdx.x & 31) == 0) red[w] = v;
  __syncthreads();
  if (threadIdx.x == 0) {
    float t = 0.f;
    for (int i = 0; i < 8; ++i) t += red[i];
    total = t;
  }
  __syncthreads();
  float r = total;
  __syncthreads();
  return r;
}

// ---------------- fp32 -> bf16 conversion ----------------
__global__ __launch_bounds__(256) void cvt_kernel(
    const float* __restrict__ src, unsigned short* __restrict__ dst, int n)
{
  int i = (blockIdx.x * 256 + threadIdx.x) * 4;
  for (int j = 0; j < 4; ++j)
    if (i + j < n) dst[i + j] = f2bf(src[i + j]);
}

// ---------------- bf16 NT GEMM: C[M,N] = A[M,K] * W[N,K]^T ----------------
// flags: 1 = +bias[n], 2 = +R residual (fp32), 4 = exact GELU, 8 = bf16 output
__global__ __launch_bounds__(256) void gemm_bf16_kernel(
    const unsigned short* __restrict__ A, const unsigned short* __restrict__ W,
    const float* __restrict__ bias, const float* __restrict__ R,
    float* __restrict__ Cf, unsigned short* __restrict__ Cb,
    int M, int N, int K, int flags)
{
  __shared__ __attribute__((aligned(16))) unsigned short As[64][32];
  __shared__ __attribute__((aligned(16))) unsigned short Ws[128][32];
  const int bm = blockIdx.x * 64;
  const int bn = blockIdx.y * 128;
  const int tid = threadIdx.x;
  const int wave = tid >> 5;
  const int wm = (wave & 1) * 32;
  const int wn = (wave >> 1) * 32;

  CFrag c[2][2];
  for (int i = 0; i < 2; ++i)
    for (int j = 0; j < 2; ++j)
      for (int r = 0; r < 8; ++r) c[i][j].f[r] = 0.f;

  for (int k0 = 0; k0 < K; k0 += 32) {
#if USE_TDM
    if (wave == 0) {
      // A tile: 64 rows x 32 k (zero-filled past M), W tile: 128 rows x 32 k
      tdm_load_2d((unsigned)(size_t)&As[0][0], A + (size_t)bm * K + k0,
                  (unsigned)(K - k0), (unsigned)(M - bm), 32u, 64u,
                  (unsigned long long)K);
      tdm_load_2d((unsigned)(size_t)&Ws[0][0], W + (size_t)bn * K + k0,
                  (unsigned)(K - k0), (unsigned)(N - bn), 32u, 128u,
                  (unsigned long long)K);
      __builtin_amdgcn_s_wait_tensorcnt(0);
      asm volatile("" ::: "memory");
    }
#else
    {
      int r = tid >> 2, seg = (tid & 3) * 8;
      u32x4 av = {0, 0, 0, 0};
      if (bm + r < M) av = *(const u32x4*)(A + (size_t)(bm + r) * K + k0 + seg);
      *(u32x4*)&As[r][seg] = av;
      for (int j = 0; j < 2; ++j) {
        int ii = tid + j * 256;
        int rr = ii >> 2, sg = (ii & 3) * 8;
        u32x4 wv = {0, 0, 0, 0};
        if (bn + rr < N) wv = *(const u32x4*)(W + (size_t)(bn + rr) * K + k0 + sg);
        *(u32x4*)&Ws[rr][sg] = wv;
      }
    }
#endif
    __syncthreads();
    v16bf a0 = load_frag16(&As[0][0], wm, 32);
    v16bf a1 = load_frag16(&As[0][0], wm + 16, 32);
    v16bf b0 = load_frag16(&Ws[0][0], wn, 32);
    v16bf b1 = load_frag16(&Ws[0][0], wn + 16, 32);
    c[0][0].v = wmma_bf16(a0, b0, c[0][0].v);
    c[0][1].v = wmma_bf16(a0, b1, c[0][1].v);
    c[1][0].v = wmma_bf16(a1, b0, c[1][0].v);
    c[1][1].v = wmma_bf16(a1, b1, c[1][1].v);
    __syncthreads();
  }

  const int lane = tid & 31;
  const int g = lane >> 4, n = lane & 15;
  for (int mi = 0; mi < 2; ++mi)
    for (int ni = 0; ni < 2; ++ni)
      for (int r = 0; r < 8; ++r) {
        int row = bm + wm + mi * 16 + r + 8 * g;
        int col = bn + wn + ni * 16 + n;
        if (row < M && col < N) {
          float v = c[mi][ni].f[r];
          if (flags & 1) v += bias[col];
          if (flags & 4) v = 0.5f * v * (1.f + erff(v * 0.70710678118654752f));
          if (flags & 2) v += R[(size_t)row * N + col];
          if (flags & 8) Cb[(size_t)row * N + col] = f2bf(v);
          else           Cf[(size_t)row * N + col] = v;
        }
      }
}

// ---------------- LayerNorm: fp32 in, bf16 out ----------------
__global__ __launch_bounds__(256) void ln_kernel(
    const float* __restrict__ x, const float* __restrict__ gw,
    unsigned short* __restrict__ y)
{
  size_t row = blockIdx.x;
  const float* xr = x + row * DD;
  float s = 0.f;
  for (int i = threadIdx.x; i < DD; i += 256) s += xr[i];
  float mean = block_sum256(s) * (1.f / DD);
  float vv = 0.f;
  for (int i = threadIdx.x; i < DD; i += 256) { float d = xr[i] - mean; vv += d * d; }
  float var = block_sum256(vv) * (1.f / DD);
  float inv = rsqrtf(var + 1e-5f);
  for (int i = threadIdx.x; i < DD; i += 256)
    y[row * DD + i] = f2bf((xr[i] - mean) * inv * gw[i]);
}

// ---------------- embedding assembly (fp32 residual stream) ----------------
__device__ inline float sinu_pos_f(int pos, int d) {
  int dd = (d < 384) ? d : d - 384;
  float f = expf(-((float)dd / 384.f) * 9.2103403719761836f);  // 10000^{-dd/384}
  float ang = (float)pos * f;
  return (d < 384) ? sinf(ang) : cosf(ang);
}

__global__ __launch_bounds__(256) void emb_kernel(
    const float* __restrict__ audio_h, const int* __restrict__ alen, const int* __restrict__ llen,
    const int* __restrict__ labels, const float* __restrict__ text_emb_w,
    const float* __restrict__ start_emb, const float* __restrict__ aps,
    const float* __restrict__ tps, float* __restrict__ x)
{
  size_t idx = (size_t)blockIdx.x * 256 + threadIdx.x;
  if (idx >= (size_t)BB * SS * DD) return;
  int d = (int)(idx % DD);
  int s = (int)((idx / DD) % SS);
  int b = (int)(idx / ((size_t)DD * SS));
  int al = alen[b], ll = llen[b];
  float v = 0.f;
  if (s < al) {
    v = audio_h[((size_t)b * TAUD + s) * DD + d] + sinu_pos_f(s, d) * aps[0];
  } else if (s == al) {
    v = start_emb[d];
  } else if (s < al + ll) {
    int tj = s - al - 1;
    if (tj < 0) tj = 0;
    if (tj > LL_ - 1) tj = LL_ - 1;
    int lab = labels[b * LL_ + tj];
    v = text_emb_w[(size_t)lab * DD + d] + sinu_pos_f(tj, d) * tps[0];
  }
  x[idx] = v;
}

// ---------------- per-wave flash attention (bf16 qkv in, bf16 out) ----------
__global__ __launch_bounds__(256) void attn_kernel(
    const unsigned short* __restrict__ qkv, unsigned short* __restrict__ o)
{
  __shared__ __attribute__((aligned(16))) unsigned short Pl[8][16][32];
  int bid = blockIdx.x;
  int qb = bid % NQB;
  int hh = (bid / NQB) % HH;
  int b  = bid / (NQB * HH);
  int wave = threadIdx.x >> 5;
  int qt = qb * QT_PER_BLK + wave;
  if (qt >= QT_PER_BH) return;
  int q0 = qt * 16;
  int lane = threadIdx.x & 31;
  int g = lane >> 4, mn = lane & 15;
  unsigned short (*P)[32] = Pl[wave];

  // Q fragments (16 x 64 -> two 16x32 A fragments) straight from global bf16
  const unsigned short* Qbase = qkv + (size_t)(b * SS + q0) * (3 * DD) + hh * DH;
  HFrag qa[2];
  qa[0].v = load_frag16(Qbase, 0, 3 * DD);
  qa[1].v = load_frag16(Qbase + 32, 0, 3 * DD);

  CFrag oc[4];
  for (int nt = 0; nt < 4; ++nt)
    for (int r = 0; r < 8; ++r) oc[nt].f[r] = 0.f;
  float rmax[8], rsum[8];
  for (int r = 0; r < 8; ++r) { rmax[r] = -1e30f; rsum[r] = 0.f; }

  const int kend = q0 + 16;
  for (int kb = 0; kb < kend; kb += 32) {
    CFrag sc[2];
    for (int st = 0; st < 2; ++st) {
      for (int r = 0; r < 8; ++r) sc[st].f[r] = 0.f;
      const unsigned short* Kbase =
          qkv + (size_t)(b * SS + kb + st * 16) * (3 * DD) + DD + hh * DH;
      v16bf k0f = load_frag16(Kbase, 0, 3 * DD);
      v16bf k1f = load_frag16(Kbase + 32, 0, 3 * DD);
      sc[st].v = wmma_bf16(qa[0].v, k0f, sc[st].v);
      sc[st].v = wmma_bf16(qa[1].v, k1f, sc[st].v);
    }
    // scale, causal mask, online softmax (row = r + 8*g; col = lane&15)
    for (int r = 0; r < 8; ++r) {
      int qpos = q0 + r + 8 * g;
      float v0 = sc[0].f[r] * 0.125f;
      float v1 = sc[1].f[r] * 0.125f;
      if (kb + mn > qpos)      v0 = -1e9f;
      if (kb + 16 + mn > qpos) v1 = -1e9f;
      float tmax = fmaxf(v0, v1);
      for (int msk = 1; msk < 16; msk <<= 1) tmax = fmaxf(tmax, __shfl_xor(tmax, msk, 32));
      float nm = fmaxf(rmax[r], tmax);
      float alpha = expf(rmax[r] - nm);
      rmax[r] = nm;
      float p0 = expf(v0 - nm), p1 = expf(v1 - nm);
      float ps = p0 + p1;
      for (int msk = 1; msk < 16; msk <<= 1) ps += __shfl_xor(ps, msk, 32);
      rsum[r] = rsum[r] * alpha + ps;
      for (int nt = 0; nt < 4; ++nt) oc[nt].f[r] *= alpha;
      P[r + 8 * g][mn]      = f2bf(p0);
      P[r + 8 * g][16 + mn] = f2bf(p1);
    }
    // re-layout P (D-frag -> A-frag) through per-wave LDS
    asm volatile("s_wait_dscnt 0" ::: "memory");
    v16bf pa = load_frag16(&P[0][0], 0, 32);
    for (int nt = 0; nt < 4; ++nt) {
      HFrag vf;
      for (int e = 0; e < 16; ++e) {
        int kk = (e < 8) ? (g * 8 + e) : (8 + g * 8 + e);
        vf.h[e] = qkv[(size_t)(b * SS + kb + kk) * (3 * DD) + 2 * DD + hh * DH + nt * 16 + mn];
      }
      oc[nt].v = wmma_bf16(pa, vf.v, oc[nt].v);
    }
  }

  for (int nt = 0; nt < 4; ++nt)
    for (int r = 0; r < 8; ++r) {
      int row = q0 + r + 8 * g;
      int col = hh * DH + nt * 16 + mn;
      o[(size_t)(b * SS + row) * DD + col] = f2bf(oc[nt].f[r] / rsum[r]);
    }
}

// ---------------- latent gather (bf16 -> bf16) ----------------
__global__ __launch_bounds__(256) void gather_kernel(
    const unsigned short* __restrict__ xf, const int* __restrict__ alen,
    const int* __restrict__ llen, unsigned short* __restrict__ lat)
{
  int idx = blockIdx.x * 256 + threadIdx.x;
  if (idx >= BB * LL_ * DD) return;
  int d = idx % DD;
  int j = (idx / DD) % LL_;
  int b = idx / (DD * LL_);
  int al = alen[b], ll = llen[b];
  int gi = al + j;
  if (gi < 0) gi = 0;
  if (gi > SS - 1) gi = SS - 1;
  lat[idx] = (j < ll) ? xf[((size_t)b * SS + gi) * DD + d] : (unsigned short)0;
}

// ---------------- per-token NLL (V=128, 128 threads) ----------------
__global__ __launch_bounds__(128) void nll_kernel(
    const float* __restrict__ logits, const int* __restrict__ labels,
    float* __restrict__ nlla, float* __restrict__ marr)
{
  int row = blockIdx.x;            // b*L + j
  int tid = threadIdx.x;
  float lg = logits[(size_t)row * VV + tid];
  float mx = lg;
  for (int m = 1; m < 32; m <<= 1) mx = fmaxf(mx, __shfl_xor(mx, m, 32));
  __shared__ float redm[4], reds[4];
  if ((tid & 31) == 0) redm[tid >> 5] = mx;
  __syncthreads();
  mx = fmaxf(fmaxf(redm[0], redm[1]), fmaxf(redm[2], redm[3]));
  float e = expf(lg - mx);
  float sm = e;
  for (int m = 1; m < 32; m <<= 1) sm += __shfl_xor(sm, m, 32);
  if ((tid & 31) == 0) reds[tid >> 5] = sm;
  __syncthreads();
  sm = reds[0] + reds[1] + reds[2] + reds[3];
  if (tid == 0) {
    float lse = logf(sm) + mx;
    int lab = labels[row];
    float nll = lse - logits[(size_t)row * VV + lab];
    float m = (lab != 0) ? 1.f : 0.f;
    nlla[row] = nll * m;
    marr[row] = m;
  }
}

__global__ __launch_bounds__(256) void reduce_kernel(
    const float* __restrict__ nlla, const float* __restrict__ marr,
    float* __restrict__ out, int n)
{
  float s = 0.f, c = 0.f;
  for (int i = threadIdx.x; i < n; i += 256) { s += nlla[i]; c += marr[i]; }
  float ts = block_sum256(s);
  float tc = block_sum256(c);
  if (threadIdx.x == 0) out[0] = ts / tc;
}

// ---------------- host side ----------------
static inline void launch_cvt(const float* s, unsigned short* d, int n, hipStream_t st) {
  cvt_kernel<<<(n + 1023) / 1024, 256, 0, st>>>(s, d, n);
}
static inline void launch_gemm(const unsigned short* A, const unsigned short* W,
                               const float* bias, const float* R,
                               float* Cf, unsigned short* Cb,
                               int M, int N, int K, int flags, hipStream_t s) {
  dim3 grid((M + 63) / 64, (N + 127) / 128);
  gemm_bf16_kernel<<<grid, 256, 0, s>>>(A, W, bias, R, Cf, Cb, M, N, K, flags);
}

extern "C" void kernel_launch(void* const* d_in, const int* in_sizes, int n_in,
                              void* d_out, int out_size, void* d_ws, size_t ws_size,
                              hipStream_t stream) {
  const float* audio        = (const float*)d_in[0];
  const int*   audio_len    = (const int*)d_in[1];
  const int*   label_len    = (const int*)d_in[2];
  const int*   labels       = (const int*)d_in[3];
  const float* text_emb_w   = (const float*)d_in[4];
  const float* text_head_w  = (const float*)d_in[5];
  const float* audio_proj_w = (const float*)d_in[6];
  const float* audio_proj_b = (const float*)d_in[7];
  const float* start_emb    = (const float*)d_in[8];
  const float* aps          = (const float*)d_in[9];
  const float* tps          = (const float*)d_in[10];
  const float* ln1_g        = (const float*)d_in[11];
  const float* ln2_g        = (const float*)d_in[12];
  const float* lnf_g        = (const float*)d_in[13];
  const float* qkv_w        = (const float*)d_in[14];
  const float* attn_o_w     = (const float*)d_in[15];
  const float* fc_w         = (const float*)d_in[16];
  const float* mlp_o_w      = (const float*)d_in[17];

  // fp32 regions
  float* x       = (float*)d_ws;                          // NTOK*DD
  float* audio_h = x + (size_t)NTOK * DD;                 // B*TA*DD
  float* logits  = audio_h + (size_t)BB * TAUD * DD;      // B*L*V
  float* nlla    = logits + (size_t)BB * LL_ * VV;        // B*L
  float* marr    = nlla + BB * LL_;                       // B*L
  // bf16 regions (ushort)
  unsigned short* audio_bf = (unsigned short*)(marr + BB * LL_);
  unsigned short* wbf      = audio_bf + (size_t)BB * TAUD * LAT;   // max 4*DD*DD
  unsigned short* hbf      = wbf + (size_t)4 * DD * DD;            // NTOK*DD
  unsigned short* qkvbf    = hbf + (size_t)NTOK * DD;              // NTOK*3*DD
  unsigned short* obf      = qkvbf + (size_t)NTOK * 3 * DD;        // NTOK*DD
  unsigned short* ffnbf    = obf + (size_t)NTOK * DD;              // NTOK*4*DD
  unsigned short* latbf    = ffnbf + (size_t)NTOK * 4 * DD;        // B*L*DD

  // 1) audio projection (bf16 GEMM): (B*TA,1024) x (768,1024)^T + bias
  launch_cvt(audio, audio_bf, BB * TAUD * LAT, stream);
  launch_cvt(audio_proj_w, wbf, DD * LAT, stream);
  launch_gemm(audio_bf, wbf, audio_proj_b, nullptr, audio_h, nullptr,
              BB * TAUD, DD, LAT, /*bias*/1, stream);

  // 2) embedding assembly -> x (fp32 residual stream)
  {
    size_t total = (size_t)BB * SS * DD;
    emb_kernel<<<(int)((total + 255) / 256), 256, 0, stream>>>(
        audio_h, audio_len, label_len, labels, text_emb_w, start_emb, aps, tps, x);
  }

  // 3) decoder layers
  for (int l = 0; l < NLAY; ++l) {
    ln_kernel<<<NTOK, 256, 0, stream>>>(x, ln1_g + (size_t)l * DD, hbf);
    launch_cvt(qkv_w + (size_t)l * 3 * DD * DD, wbf, 3 * DD * DD, stream);
    launch_gemm(hbf, wbf, nullptr, nullptr, nullptr, qkvbf,
                NTOK, 3 * DD, DD, /*bf16 out*/8, stream);
    attn_kernel<<<BB * HH * NQB, 256, 0, stream>>>(qkvbf, obf);
    launch_cvt(attn_o_w + (size_t)l * DD * DD, wbf, DD * DD, stream);
    launch_gemm(obf, wbf, nullptr, x, x, nullptr,
                NTOK, DD, DD, /*residual*/2, stream);
    ln_kernel<<<NTOK, 256, 0, stream>>>(x, ln2_g + (size_t)l * DD, hbf);
    launch_cvt(fc_w + (size_t)l * 4 * DD * DD, wbf, 4 * DD * DD, stream);
    launch_gemm(hbf, wbf, nullptr, nullptr, nullptr, ffnbf,
                NTOK, 4 * DD, DD, /*gelu+bf16*/4 | 8, stream);
    launch_cvt(mlp_o_w + (size_t)l * DD * 4 * DD, wbf, DD * 4 * DD, stream);
    launch_gemm(ffnbf, wbf, nullptr, x, x, nullptr,
                NTOK, DD, 4 * DD, /*residual*/2, stream);
  }

  // 4) final LN -> hbf
  ln_kernel<<<NTOK, 256, 0, stream>>>(x, lnf_g, hbf);

  // 5) gather latents -> latbf (B*L x D)
  {
    int total = BB * LL_ * DD;
    gather_kernel<<<(total + 255) / 256, 256, 0, stream>>>(hbf, audio_len, label_len, latbf);
  }

  // 6) logits: (B*L,768) x (128,768)^T -> fp32
  launch_cvt(text_head_w, wbf, VV * DD, stream);
  launch_gemm(latbf, wbf, nullptr, nullptr, logits, nullptr,
              BB * LL_, VV, DD, 0, stream);

  // 7) per-token NLL + masked mean
  nll_kernel<<<BB * LL_, 128, 0, stream>>>(logits, labels, nlla, marr);
  reduce_kernel<<<1, 256, 0, stream>>>(nlla, marr, (float*)d_out, BB * LL_);
}
